// DecoderBlock_55087250538577
// MI455X (gfx1250) — matrix operand order repaired and checked by reading, not compiled, not run on hardware
//
#include <hip/hip_runtime.h>
#include <hip/hip_bf16.h>

typedef __attribute__((ext_vector_type(16))) _Float16 v16h;
typedef __attribute__((ext_vector_type(8)))  _Float16 v8h;
typedef __attribute__((ext_vector_type(4)))  _Float16 v4h;
typedef __attribute__((ext_vector_type(8)))  float    v8f;

#define E_  512
#define H_  8
#define D_  64
#define B_  64
#define T_  256
#define S_  256

// splice two 16B loads into a 32B WMMA fragment (register-only)
__device__ __forceinline__ v16h mkfrag(const _Float16* p0, const _Float16* p1) {
  v8h x = *(const v8h*)p0;
  v8h y = *(const v8h*)p1;
  return __builtin_shufflevector(x, y, 0,1,2,3,4,5,6,7,8,9,10,11,12,13,14,15);
}

// ---------------------------------------------------------------------------
// Pre-pass: f32 -> f16 convert (vectorized)
// ---------------------------------------------------------------------------
__global__ __launch_bounds__(256) void convert_f16_kernel(
    const float* __restrict__ s, _Float16* __restrict__ d, int n4)
{
  int u = blockIdx.x * 256 + threadIdx.x;
  if (u >= n4) return;
  float4 v = ((const float4*)s)[u];
  v4h t; t[0]=(_Float16)v.x; t[1]=(_Float16)v.y; t[2]=(_Float16)v.z; t[3]=(_Float16)v.w;
  ((v4h*)d)[u] = t;
}

// ---------------------------------------------------------------------------
// Pre-pass: W[K,N] f32 -> Wt[N,K] f16 (coalesced reads, b64 writes)
// ---------------------------------------------------------------------------
__global__ __launch_bounds__(256) void transpose_w_kernel(
    const float* __restrict__ W, _Float16* __restrict__ Wt, int N, int K, int lgN)
{
  int u = blockIdx.x * 256 + threadIdx.x;        // over N*K/4 units
  if (u >= (N * K) >> 2) return;
  int n  = u & (N - 1);
  int k4 = (u >> lgN) << 2;
  const float* wp = &W[(size_t)k4 * N + n];
  float w0 = wp[0];
  float w1 = wp[N];
  float w2 = wp[2 * (size_t)N];
  float w3 = wp[3 * (size_t)N];
  v4h t; t[0]=(_Float16)w0; t[1]=(_Float16)w1; t[2]=(_Float16)w2; t[3]=(_Float16)w3;
  *(v4h*)&Wt[(size_t)n * K + k4] = t;
}

// ---------------------------------------------------------------------------
// LDS-free WMMA GEMM:  C[M,N] = A[M,K](f16) * Wt[N,K](f16)^T (+bias)
// Block 64x256, 8 waves (2 mi x 4 ni); wave tile 32x64 -> 8 WMMAs / 32-K chunk,
// with each B fragment reused by both A row-blocks (1.5 loads per WMMA).
// Output: f32 (Cf) or f16 row-major (Ch) or f16 transposed [B,H,D,S] (Ch,transV)
// ---------------------------------------------------------------------------
#define BM 64
#define BN 256

__global__ __launch_bounds__(256) void gemm_f16_kernel(
    const _Float16* __restrict__ A, const _Float16* __restrict__ Wt,
    const float* __restrict__ bias, float* __restrict__ Cf,
    _Float16* __restrict__ Ch,
    int M, int N, int K, int relu, int transV)
{
  const int lane = threadIdx.x & 31;
  const int wave = threadIdx.x >> 5;
  const int mi   = wave & 1;    // 32-row half
  const int ni   = wave >> 1;   // 64-col quarter
  const int lo   = lane & 15;
  const int hi   = lane >> 4;

  const int m0 = blockIdx.x * BM;
  const int n0 = blockIdx.y * BN;

  const _Float16* a0r = A + (size_t)(m0 + mi * 32 + lo) * K;
  const _Float16* a1r = a0r + (size_t)16 * K;
  const _Float16* w0r = Wt + (size_t)(n0 + ni * 64 +  0 + lo) * K;
  const _Float16* w1r = Wt + (size_t)(n0 + ni * 64 + 16 + lo) * K;
  const _Float16* w2r = Wt + (size_t)(n0 + ni * 64 + 32 + lo) * K;
  const _Float16* w3r = Wt + (size_t)(n0 + ni * 64 + 48 + lo) * K;

  v8f acc0[4] = {{}, {}, {}, {}};
  v8f acc1[4] = {{}, {}, {}, {}};

  #pragma unroll 2
  for (int k0 = 0; k0 < K; k0 += 32) {
    v16h af0 = mkfrag(&a0r[k0 + hi * 16], &a0r[k0 + hi * 16 + 8]);
    v16h af1 = mkfrag(&a1r[k0 + hi * 16], &a1r[k0 + hi * 16 + 8]);
    v16h b0  = mkfrag(&w0r[k0 + hi * 16], &w0r[k0 + hi * 16 + 8]);
    v16h b1  = mkfrag(&w1r[k0 + hi * 16], &w1r[k0 + hi * 16 + 8]);
    v16h b2  = mkfrag(&w2r[k0 + hi * 16], &w2r[k0 + hi * 16 + 8]);
    v16h b3  = mkfrag(&w3r[k0 + hi * 16], &w3r[k0 + hi * 16 + 8]);
    acc0[0] = __builtin_amdgcn_wmma_f32_16x16x32_f16(false, af0, false, b0, (short)0, acc0[0], false, false);
    acc1[0] = __builtin_amdgcn_wmma_f32_16x16x32_f16(false, af1, false, b0, (short)0, acc1[0], false, false);
    acc0[1] = __builtin_amdgcn_wmma_f32_16x16x32_f16(false, af0, false, b1, (short)0, acc0[1], false, false);
    acc1[1] = __builtin_amdgcn_wmma_f32_16x16x32_f16(false, af1, false, b1, (short)0, acc1[1], false, false);
    acc0[2] = __builtin_amdgcn_wmma_f32_16x16x32_f16(false, af0, false, b2, (short)0, acc0[2], false, false);
    acc1[2] = __builtin_amdgcn_wmma_f32_16x16x32_f16(false, af1, false, b2, (short)0, acc1[2], false, false);
    acc0[3] = __builtin_amdgcn_wmma_f32_16x16x32_f16(false, af0, false, b3, (short)0, acc0[3], false, false);
    acc1[3] = __builtin_amdgcn_wmma_f32_16x16x32_f16(false, af1, false, b3, (short)0, acc1[3], false, false);
  }

  if (transV) {
    // write f16 transposed [B,H,D,S]: 8 consecutive s per lane -> packed b128
    #pragma unroll
    for (int g = 0; g < 2; ++g) {
      const int row0 = m0 + mi * 32 + g * 16 + hi * 8;
      const int bb = row0 >> 8;          // T_==S_==256
      const int s0 = row0 & 255;
      #pragma unroll
      for (int f = 0; f < 4; ++f) {
        int col = n0 + ni * 64 + f * 16 + lo;
        float bv = bias ? bias[col] : 0.0f;
        int hh = col >> 6, d = col & 63;
        const v8f& a = g ? acc1[f] : acc0[f];
        v8h pack;
        #pragma unroll
        for (int j = 0; j < 8; ++j) pack[j] = (_Float16)(a[j] + bv);
        *(v8h*)&Ch[(((size_t)bb * H_ + hh) * D_ + d) * S_ + s0] = pack;
      }
    }
    return;
  }
  #pragma unroll
  for (int g = 0; g < 2; ++g) {
    #pragma unroll
    for (int f = 0; f < 4; ++f) {
      int cb = n0 + ni * 64 + f * 16 + lo;
      float bv = bias ? bias[cb] : 0.0f;
      const v8f& a = g ? acc1[f] : acc0[f];
      #pragma unroll
      for (int j = 0; j < 8; ++j) {
        int row = m0 + mi * 32 + g * 16 + hi * 8 + j;
        float v = a[j] + bv;
        if (relu) v = fmaxf(v, 0.0f);
        if (Cf) Cf[(size_t)row * N + cb] = v;
        if (Ch) Ch[(size_t)row * N + cb] = (_Float16)v;
      }
    }
  }
}

// ---------------------------------------------------------------------------
// Flash attention, fully global-fed fragments; one wave per 32 query rows.
// Q,K: [B,T,H,D] f16 row-major; Vt: [B,H,D,S] f16; O: [B,T,H,D] f16.
// K/V fragments reused across both 16-row Q blocks (16 WMMAs per 32-col chunk).
// Only P (softmax weights) round-trips through LDS for the C->A relayout.
// ---------------------------------------------------------------------------
#define PH_W 40
#define PH_SZ (32 * PH_W)

__global__ __launch_bounds__(128) void attn_kernel(
    const _Float16* __restrict__ Qm, const _Float16* __restrict__ Km,
    const _Float16* __restrict__ Vt, _Float16* __restrict__ Om,
    int Tq, int Tk, int causal)
{
  __shared__ alignas(16) _Float16 PhS[4 * PH_SZ];

  const int lane = threadIdx.x & 31;
  const int wid  = threadIdx.x >> 5;
  const int gw   = blockIdx.x * 4 + wid;
  const int nTb  = Tq >> 5;
  const int tb   = gw % nTb;
  const int bh   = gw / nTb;
  if (bh >= B_ * H_) return;
  const int h  = bh % H_;
  const int b  = bh / H_;
  const int lo = lane & 15;
  const int hi = lane >> 4;
  const int t0 = tb << 5;

  _Float16* ph = PhS + wid * PH_SZ;

  // Q fragments for two 16-row blocks: direct global b128 loads
  const _Float16* qrA = Qm + ((size_t)(b * Tq + t0 + lo) * H_ + h) * D_;
  const _Float16* qrB = qrA + (size_t)16 * H_ * D_;
  v16h qf0a = mkfrag(&qrA[16 * hi], &qrA[16 * hi + 8]);
  v16h qf1a = mkfrag(&qrA[32 + 16 * hi], &qrA[32 + 16 * hi + 8]);
  v16h qf0b = mkfrag(&qrB[16 * hi], &qrB[16 * hi + 8]);
  v16h qf1b = mkfrag(&qrB[32 + 16 * hi], &qrB[32 + 16 * hi + 8]);

  const _Float16* vbase = Vt + ((size_t)(b * H_ + h) * D_) * Tk;

  float mA[8], lA[8], mB[8], lB[8];
  v8f oA0 = {}, oA1 = {}, oA2 = {}, oA3 = {};
  v8f oB0 = {}, oB1 = {}, oB2 = {}, oB3 = {};
  #pragma unroll
  for (int j = 0; j < 8; ++j) {
    mA[j] = -__builtin_inff(); lA[j] = 0.0f;
    mB[j] = -__builtin_inff(); lB[j] = 0.0f;
  }

  const int send = causal ? (t0 + 32) : Tk;
  for (int sc = 0; sc < send; sc += 32) {
    // ---- S = Q * K^T : K^T fragments contiguous in global, reused by A & B
    const _Float16* kr0 = Km + ((size_t)(b * Tk + sc + lo) * H_ + h) * D_;
    const _Float16* kr1 = kr0 + (size_t)16 * H_ * D_;
    v16h k00 = mkfrag(&kr0[hi * 16],      &kr0[hi * 16 + 8]);
    v16h k01 = mkfrag(&kr0[32 + hi * 16], &kr0[32 + hi * 16 + 8]);
    v16h k10 = mkfrag(&kr1[hi * 16],      &kr1[hi * 16 + 8]);
    v16h k11 = mkfrag(&kr1[32 + hi * 16], &kr1[32 + hi * 16 + 8]);
    v8f sA0 = {}, sA1 = {}, sB0 = {}, sB1 = {};
    sA0 = __builtin_amdgcn_wmma_f32_16x16x32_f16(false, qf0a, false, k00, (short)0, sA0, false, false);
    sA0 = __builtin_amdgcn_wmma_f32_16x16x32_f16(false, qf1a, false, k01, (short)0, sA0, false, false);
    sA1 = __builtin_amdgcn_wmma_f32_16x16x32_f16(false, qf0a, false, k10, (short)0, sA1, false, false);
    sA1 = __builtin_amdgcn_wmma_f32_16x16x32_f16(false, qf1a, false, k11, (short)0, sA1, false, false);
    sB0 = __builtin_amdgcn_wmma_f32_16x16x32_f16(false, qf0b, false, k00, (short)0, sB0, false, false);
    sB0 = __builtin_amdgcn_wmma_f32_16x16x32_f16(false, qf1b, false, k01, (short)0, sB0, false, false);
    sB1 = __builtin_amdgcn_wmma_f32_16x16x32_f16(false, qf0b, false, k10, (short)0, sB1, false, false);
    sB1 = __builtin_amdgcn_wmma_f32_16x16x32_f16(false, qf1b, false, k11, (short)0, sB1, false, false);

    // ---- scale + causal mask + online softmax, per row block
    float pA0[8], pA1[8], pB0[8], pB1[8], mcA[8], mcB[8];
    #pragma unroll
    for (int j = 0; j < 8; ++j) {
      int rowA = t0 + hi * 8 + j;
      int rowB = rowA + 16;
      float a0v = sA0[j] * 0.125f, a1v = sA1[j] * 0.125f;
      float b0v = sB0[j] * 0.125f, b1v = sB1[j] * 0.125f;
      if (causal) {
        if (sc + lo      > rowA) a0v = -1e30f;
        if (sc + 16 + lo > rowA) a1v = -1e30f;
        if (sc + lo      > rowB) b0v = -1e30f;
        if (sc + 16 + lo > rowB) b1v = -1e30f;
      }
      pA0[j] = a0v; pA1[j] = a1v; mcA[j] = fmaxf(a0v, a1v);
      pB0[j] = b0v; pB1[j] = b1v; mcB[j] = fmaxf(b0v, b1v);
    }
    #pragma unroll
    for (int j = 0; j < 8; ++j) {
      #pragma unroll
      for (int msk = 1; msk < 16; msk <<= 1) {
        mcA[j] = fmaxf(mcA[j], __shfl_xor(mcA[j], msk, 16));
        mcB[j] = fmaxf(mcB[j], __shfl_xor(mcB[j], msk, 16));
      }
    }
    float alA[8], alB[8];
    #pragma unroll
    for (int j = 0; j < 8; ++j) {
      float mnA = fmaxf(mA[j], mcA[j]);
      float mnB = fmaxf(mB[j], mcB[j]);
      alA[j] = __expf(mA[j] - mnA);
      alB[j] = __expf(mB[j] - mnB);
      mA[j] = mnA; mB[j] = mnB;
      pA0[j] = __expf(pA0[j] - mnA); pA1[j] = __expf(pA1[j] - mnA);
      pB0[j] = __expf(pB0[j] - mnB); pB1[j] = __expf(pB1[j] - mnB);
      float lsA = pA0[j] + pA1[j];
      float lsB = pB0[j] + pB1[j];
      #pragma unroll
      for (int msk = 1; msk < 16; msk <<= 1) {
        lsA += __shfl_xor(lsA, msk, 16);
        lsB += __shfl_xor(lsB, msk, 16);
      }
      lA[j] = lA[j] * alA[j] + lsA;
      lB[j] = lB[j] * alB[j] + lsB;
    }
    #pragma unroll
    for (int j = 0; j < 8; ++j) {
      oA0[j] *= alA[j]; oA1[j] *= alA[j]; oA2[j] *= alA[j]; oA3[j] *= alA[j];
      oB0[j] *= alB[j]; oB1[j] *= alB[j]; oB2[j] *= alB[j]; oB3[j] *= alB[j];
    }

    // ---- P: C layout -> LDS -> A fragments (rows 0..15 = block A, 16..31 = B)
    #pragma unroll
    for (int j = 0; j < 8; ++j) {
      int prA = hi * 8 + j;
      int prB = 16 + prA;
      ph[prA * PH_W + lo]      = (_Float16)pA0[j];
      ph[prA * PH_W + 16 + lo] = (_Float16)pA1[j];
      ph[prB * PH_W + lo]      = (_Float16)pB0[j];
      ph[prB * PH_W + 16 + lo] = (_Float16)pB1[j];
    }
    asm volatile("s_wait_dscnt 0x0" ::: "memory");
    v16h pfA = mkfrag(&ph[lo * PH_W + hi * 16],        &ph[lo * PH_W + hi * 16 + 8]);
    v16h pfB = mkfrag(&ph[(16 + lo) * PH_W + hi * 16], &ph[(16 + lo) * PH_W + hi * 16 + 8]);

    // ---- O += P * V : V^T fragments contiguous in global, reused by A & B
    const _Float16* vc = vbase + sc + hi * 16;
    v16h vf;
    vf = mkfrag(&vc[(     lo) * Tk], &vc[(     lo) * Tk + 8]);
    oA0 = __builtin_amdgcn_wmma_f32_16x16x32_f16(false, pfA, false, vf, (short)0, oA0, false, false);
    oB0 = __builtin_amdgcn_wmma_f32_16x16x32_f16(false, pfB, false, vf, (short)0, oB0, false, false);
    vf = mkfrag(&vc[(16 + lo) * Tk], &vc[(16 + lo) * Tk + 8]);
    oA1 = __builtin_amdgcn_wmma_f32_16x16x32_f16(false, pfA, false, vf, (short)0, oA1, false, false);
    oB1 = __builtin_amdgcn_wmma_f32_16x16x32_f16(false, pfB, false, vf, (short)0, oB1, false, false);
    vf = mkfrag(&vc[(32 + lo) * Tk], &vc[(32 + lo) * Tk + 8]);
    oA2 = __builtin_amdgcn_wmma_f32_16x16x32_f16(false, pfA, false, vf, (short)0, oA2, false, false);
    oB2 = __builtin_amdgcn_wmma_f32_16x16x32_f16(false, pfB, false, vf, (short)0, oB2, false, false);
    vf = mkfrag(&vc[(48 + lo) * Tk], &vc[(48 + lo) * Tk + 8]);
    oA3 = __builtin_amdgcn_wmma_f32_16x16x32_f16(false, pfA, false, vf, (short)0, oA3, false, false);
    oB3 = __builtin_amdgcn_wmma_f32_16x16x32_f16(false, pfB, false, vf, (short)0, oB3, false, false);
    asm volatile("s_wait_dscnt 0x0" ::: "memory");  // LDS reads done before next P store
  }

  // normalize and write O [B,T,H,D] f16
  #pragma unroll
  for (int j = 0; j < 8; ++j) {
    float invA = 1.0f / lA[j];
    float invB = 1.0f / lB[j];
    int rowA = t0 + hi * 8 + j;
    _Float16* orA = Om + ((size_t)(b * Tq + rowA) * H_ + h) * D_;
    _Float16* orB = orA + (size_t)16 * H_ * D_;
    orA[lo]      = (_Float16)(oA0[j] * invA);
    orA[16 + lo] = (_Float16)(oA1[j] * invA);
    orA[32 + lo] = (_Float16)(oA2[j] * invA);
    orA[48 + lo] = (_Float16)(oA3[j] * invA);
    orB[lo]      = (_Float16)(oB0[j] * invB);
    orB[16 + lo] = (_Float16)(oB1[j] * invB);
    orB[32 + lo] = (_Float16)(oB2[j] * invB);
    orB[48 + lo] = (_Float16)(oB3[j] * invB);
  }
}

// ---------------------------------------------------------------------------
// Y = LayerNorm(X + R) * g + b  (wave per row); optional f16 mirror Y16
// ---------------------------------------------------------------------------
__global__ __launch_bounds__(256) void add_ln_kernel(
    const float* __restrict__ X, const float* __restrict__ R,
    const float* __restrict__ g, const float* __restrict__ bta,
    float* __restrict__ Y, _Float16* __restrict__ Y16, int rows)
{
  int gw   = (int)(((size_t)blockIdx.x * blockDim.x + threadIdx.x) >> 5);
  int lane = threadIdx.x & 31;
  if (gw >= rows) return;
  const float* xr = X + (size_t)gw * E_;
  const float* rr = R + (size_t)gw * E_;
  float v[16];
  float s = 0.0f;
  #pragma unroll
  for (int i = 0; i < 16; ++i) {
    v[i] = xr[lane + 32 * i] + rr[lane + 32 * i];
    s += v[i];
  }
  #pragma unroll
  for (int msk = 1; msk < 32; msk <<= 1) s += __shfl_xor(s, msk, 32);
  float mean = s * (1.0f / E_);
  float s2 = 0.0f;
  #pragma unroll
  for (int i = 0; i < 16; ++i) { float d = v[i] - mean; s2 += d * d; }
  #pragma unroll
  for (int msk = 1; msk < 32; msk <<= 1) s2 += __shfl_xor(s2, msk, 32);
  float rstd = rsqrtf(s2 * (1.0f / E_) + 1e-5f);
  float* yr = Y + (size_t)gw * E_;
  _Float16* hr = Y16 ? Y16 + (size_t)gw * E_ : nullptr;
  #pragma unroll
  for (int i = 0; i < 16; ++i) {
    int c = lane + 32 * i;
    float o = (v[i] - mean) * rstd * g[c] + bta[c];
    yr[c] = o;
    if (hr) hr[c] = (_Float16)o;
  }
}

// ---------------------------------------------------------------------------
extern "C" void kernel_launch(void* const* d_in, const int* in_sizes, int n_in,
                              void* d_out, int out_size, void* d_ws, size_t ws_size,
                              hipStream_t stream) {
  (void)in_sizes; (void)n_in; (void)out_size; (void)ws_size;
  const float* x     = (const float*)d_in[0];
  const float* enc   = (const float*)d_in[1];
  const float* mq_w  = (const float*)d_in[2];
  const float* mk_w  = (const float*)d_in[3];
  const float* mv_w  = (const float*)d_in[4];
  const float* mpj_w = (const float*)d_in[5];
  const float* mpj_b = (const float*)d_in[6];
  const float* cq_w  = (const float*)d_in[7];
  const float* cq_b  = (const float*)d_in[8];
  const float* ck_w  = (const float*)d_in[9];
  const float* ck_b  = (const float*)d_in[10];
  const float* cv_w  = (const float*)d_in[11];
  const float* cv_b  = (const float*)d_in[12];
  const float* co_w  = (const float*)d_in[13];
  const float* co_b  = (const float*)d_in[14];
  const float* f1_w  = (const float*)d_in[15];
  const float* f1_b  = (const float*)d_in[16];
  const float* f2_w  = (const float*)d_in[17];
  const float* f2_b  = (const float*)d_in[18];
  const float* ln1_s = (const float*)d_in[19];
  const float* ln1_b = (const float*)d_in[20];
  const float* ln2_s = (const float*)d_in[21];
  const float* ln2_b = (const float*)d_in[22];
  const float* ln3_s = (const float*)d_in[23];
  const float* ln3_b = (const float*)d_in[24];

  const int M = B_ * T_;              // 16384 (== B_*S_)
  const size_t SZ = (size_t)M * E_;   // elements per [M,512]

  // ---- workspace carve-up (16B aligned everywhere; sizes are pow2 multiples)
  char* p = (char*)d_ws;
  float*     tp   = (float*)p;                 p += SZ * 4;        // f32 [M,512]
  float*     x1   = (float*)p;                 p += SZ * 4;
  float*     x2   = (float*)p;                 p += SZ * 4;
  _Float16*  xh   = (_Float16*)p;              p += SZ * 2;        // f16 [M,512]
  _Float16*  ench = (_Float16*)p;              p += SZ * 2;
  _Float16*  qh   = (_Float16*)p;              p += SZ * 2;
  _Float16*  kh   = (_Float16*)p;              p += SZ * 2;
  _Float16*  vt   = (_Float16*)p;              p += SZ * 2;        // f16 [B,H,D,S]
  _Float16*  aoh  = (_Float16*)p;              p += SZ * 2;
  _Float16*  x1h  = (_Float16*)p;              p += SZ * 2;
  _Float16*  x2h  = (_Float16*)p;              p += SZ * 2;
  _Float16*  hh   = (_Float16*)p;              p += SZ * 2 * 4;    // f16 [M,2048]
  const size_t W55 = (size_t)E_ * E_;          // 512*512
  _Float16*  wq   = (_Float16*)p;              p += W55 * 2;       // all Wt: [N,K] f16
  _Float16*  wk   = (_Float16*)p;              p += W55 * 2;
  _Float16*  wv   = (_Float16*)p;              p += W55 * 2;
  _Float16*  wpj  = (_Float16*)p;              p += W55 * 2;
  _Float16*  wcq  = (_Float16*)p;              p += W55 * 2;
  _Float16*  wck  = (_Float16*)p;              p += W55 * 2;
  _Float16*  wcv  = (_Float16*)p;              p += W55 * 2;
  _Float16*  wco  = (_Float16*)p;              p += W55 * 2;
  _Float16*  wf1  = (_Float16*)p;              p += W55 * 4 * 2;   // [2048,512]
  _Float16*  wf2  = (_Float16*)p;              p += W55 * 4 * 2;   // [512,2048]

  dim3 blk(256);
  dim3 cgrd((unsigned)(SZ / 4 / 256));          // convert grids
  dim3 t55((unsigned)(W55 / 4 / 256));          // 512x512 transpose
  dim3 t5x((unsigned)(W55 * 4 / 4 / 256));      // 512x2048 / 2048x512
  dim3 g512(M / BM, E_ / BN);                   // 256 x 2
  dim3 g2048(M / BM, (4 * E_) / BN);            // 256 x 8
  dim3 ablk(128);
  dim3 agrd((B_ * H_ * (T_ / 32)) / 4);         // 1024
  dim3 lgrd(M / 8);

  // ---- pre-pass: f16 activations + transposed f16 weights
  convert_f16_kernel<<<cgrd, blk, 0, stream>>>(x,   xh,   (int)(SZ / 4));
  convert_f16_kernel<<<cgrd, blk, 0, stream>>>(enc, ench, (int)(SZ / 4));
  transpose_w_kernel<<<t55, blk, 0, stream>>>(mq_w,  wq,  E_, E_, 9);
  transpose_w_kernel<<<t55, blk, 0, stream>>>(mk_w,  wk,  E_, E_, 9);
  transpose_w_kernel<<<t55, blk, 0, stream>>>(mv_w,  wv,  E_, E_, 9);
  transpose_w_kernel<<<t55, blk, 0, stream>>>(mpj_w, wpj, E_, E_, 9);
  transpose_w_kernel<<<t55, blk, 0, stream>>>(cq_w,  wcq, E_, E_, 9);
  transpose_w_kernel<<<t55, blk, 0, stream>>>(ck_w,  wck, E_, E_, 9);
  transpose_w_kernel<<<t55, blk, 0, stream>>>(cv_w,  wcv, E_, E_, 9);
  transpose_w_kernel<<<t55, blk, 0, stream>>>(co_w,  wco, E_, E_, 9);
  transpose_w_kernel<<<t5x, blk, 0, stream>>>(f1_w,  wf1, 4 * E_, E_, 11);
  transpose_w_kernel<<<t5x, blk, 0, stream>>>(f2_w,  wf2, E_, 4 * E_, 9);

  // --- masked multi-head self-attention ---
  gemm_f16_kernel<<<g512, blk, 0, stream>>>(xh, wq, nullptr, nullptr, qh, M, E_, E_, 0, 0);
  gemm_f16_kernel<<<g512, blk, 0, stream>>>(xh, wk, nullptr, nullptr, kh, M, E_, E_, 0, 0);
  gemm_f16_kernel<<<g512, blk, 0, stream>>>(xh, wv, nullptr, nullptr, vt, M, E_, E_, 0, 1);
  attn_kernel<<<agrd, ablk, 0, stream>>>(qh, kh, vt, aoh, T_, T_, 1);
  gemm_f16_kernel<<<g512, blk, 0, stream>>>(aoh, wpj, mpj_b, tp, nullptr, M, E_, E_, 0, 0);
  add_ln_kernel<<<lgrd, blk, 0, stream>>>(x, tp, ln1_s, ln1_b, x1, x1h, M);
  // --- cross-attention ---
  gemm_f16_kernel<<<g512, blk, 0, stream>>>(x1h,  wcq, cq_b, nullptr, qh, M, E_, E_, 0, 0);
  gemm_f16_kernel<<<g512, blk, 0, stream>>>(ench, wck, ck_b, nullptr, kh, M, E_, E_, 0, 0);
  gemm_f16_kernel<<<g512, blk, 0, stream>>>(ench, wcv, cv_b, nullptr, vt, M, E_, E_, 0, 1);
  attn_kernel<<<agrd, ablk, 0, stream>>>(qh, kh, vt, aoh, T_, S_, 0);
  gemm_f16_kernel<<<g512, blk, 0, stream>>>(aoh, wco, co_b, tp, nullptr, M, E_, E_, 0, 0);
  add_ln_kernel<<<lgrd, blk, 0, stream>>>(x1, tp, ln2_s, ln2_b, x2, x2h, M);
  // --- feed-forward ---
  gemm_f16_kernel<<<g2048, blk, 0, stream>>>(x2h, wf1, f1_b, nullptr, hh, M, 4 * E_, E_, 1, 0);
  gemm_f16_kernel<<<g512,  blk, 0, stream>>>(hh,  wf2, f2_b, tp, nullptr, M, E_, 4 * E_, 0, 0);
  add_ln_kernel<<<lgrd, blk, 0, stream>>>(x2, tp, ln3_s, ln3_b, (float*)d_out, nullptr, M);
}
